// TransformerEncoderClassifier_50835232915828
// MI455X (gfx1250) — matrix operand order repaired and checked by reading, not compiled, not run on hardware
//
#include <hip/hip_runtime.h>

#define DEV __device__ __forceinline__

typedef unsigned int u32;
typedef unsigned short u16;
typedef __attribute__((ext_vector_type(16))) __bf16 v16bf;
typedef __attribute__((ext_vector_type(8)))  float  v8f;

// Model dims
enum { B_ = 8, S_ = 1024, D_ = 256, H_ = 8, L_ = 2, FF_ = 1024, G_ = 12, DH_ = 32, C_ = 4, BS_ = B_ * S_ };

union FragAB { v16bf v; u32 u[8]; };

DEV u16 f2bf(float f) {
    union { float f; u32 u; } x; x.f = f;
    u32 r = x.u + 0x7FFFu + ((x.u >> 16) & 1u);   // round-to-nearest-even
    return (u16)(r >> 16);
}

// ---------------------------------------------------------------- f32 -> bf16
__global__ void cvt_f32_bf16(const float* __restrict__ in, u16* __restrict__ out, int n) {
    int i = blockIdx.x * blockDim.x + threadIdx.x;
    if (i < n) out[i] = f2bf(in[i]);
}

// ---------------------------------------------------------------- embedding
__global__ void embed_kernel(const int* __restrict__ ids, const int* __restrict__ ngram_ids,
                             const float* __restrict__ tfidf, const float* __restrict__ emb,
                             const float* __restrict__ pos, const float* __restrict__ bucket,
                             float* __restrict__ x, float* __restrict__ tf, float* __restrict__ pm) {
    int tok = blockIdx.x;              // 0 .. B*S-1
    int d   = threadIdx.x;             // 0 .. D-1
    int id  = ids[tok];
    int s   = tok & (S_ - 1);
    float v = emb[(size_t)id * D_ + d] + pos[s * D_ + d];
    float acc = 0.f, cnt = 0.f;
    for (int g = 0; g < G_; ++g) {
        int ng = ngram_ids[id * G_ + g];
        if (ng != 0) { acc += bucket[(size_t)ng * D_ + d]; cnt += 1.f; }
    }
    v += acc / fmaxf(cnt, 1.f);
    x[(size_t)tok * D_ + d] = v;
    if (d == 0) { tf[tok] = tfidf[id]; pm[tok] = (id == 0) ? 1.f : 0.f; }
}

// ---------------------------------------------------------------- layernorm -> bf16
__global__ __launch_bounds__(256) void ln_bf16_kernel(const float* __restrict__ x,
                                                      const float* __restrict__ g,
                                                      const float* __restrict__ b,
                                                      u16* __restrict__ out) {
    int w = threadIdx.x >> 5, lane = threadIdx.x & 31;
    int row = blockIdx.x * 8 + w;
    const float* xr = x + (size_t)row * D_;
    float r[8]; float s = 0.f;
#pragma unroll
    for (int i = 0; i < 8; ++i) { r[i] = xr[lane + 32 * i]; s += r[i]; }
    for (int o = 16; o > 0; o >>= 1) s += __shfl_xor(s, o, 32);
    float m = s * (1.f / D_);
    float s2 = 0.f;
#pragma unroll
    for (int i = 0; i < 8; ++i) { float d0 = r[i] - m; s2 += d0 * d0; }
    for (int o = 16; o > 0; o >>= 1) s2 += __shfl_xor(s2, o, 32);
    float inv = rsqrtf(s2 * (1.f / D_) + 1e-5f);
    u16* orow = out + (size_t)row * D_;
#pragma unroll
    for (int i = 0; i < 8; ++i) {
        int d0 = lane + 32 * i;
        orow[d0] = f2bf((r[i] - m) * inv * g[d0] + b[d0]);
    }
}

// ---------------------------------------------------------------- WMMA GEMM
// C[M,N] = A[M,K](bf16) @ W[K,N](bf16) + bias, templated epilogue.
// Block: 256 thr (8 waves); tile 128(M) x 64(N); K step 32; software-pipelined
// register->LDS staging so global loads of tile k+1 overlap WMMAs of tile k.
template <int RELU, int HAS_RESID, int OUTF, int OUTB>
__global__ __launch_bounds__(256) void gemm_wmma(const u16* __restrict__ A, const u16* __restrict__ W,
                                                 const float* __restrict__ bias,
                                                 const float* __restrict__ resid,
                                                 float* __restrict__ outF, u16* __restrict__ outB,
                                                 int M, int N, int K) {
    __shared__ __align__(16) u32 ldsA[128 * 16];   // 128 rows x 32 bf16
    __shared__ __align__(16) u32 ldsB[32 * 32];    // 32 rows  x 64 bf16
    int tid = threadIdx.x, w = tid >> 5, lane = tid & 31;
    int half = lane >> 4, l16 = lane & 15;
    int m0 = blockIdx.y * 128, n0 = blockIdx.x * 64;

    int rowA = tid >> 2, qA = tid & 3;             // A tile: 2 uint4 per thread
    int rowB = tid >> 3, qB = tid & 7;             // B tile: 1 uint4 per thread

    v8f acc[4];
#pragma unroll
    for (int nt = 0; nt < 4; ++nt)
#pragma unroll
        for (int v = 0; v < 8; ++v) acc[nt][v] = 0.f;

    uint4 ra0, ra1, rb;
    auto gload = [&](int k0) {
        ra0 = *((const uint4*)(A + (size_t)(m0 + rowA) * K + k0) + qA);
        ra1 = *((const uint4*)(A + (size_t)(m0 + rowA + 64) * K + k0) + qA);
        rb  = *((const uint4*)(W + (size_t)(k0 + rowB) * N + n0) + qB);
    };
    gload(0);

    for (int k0 = 0; k0 < K; k0 += 32) {
        ((uint4*)ldsA)[tid]       = ra0;
        ((uint4*)ldsA)[tid + 256] = ra1;
        ((uint4*)ldsB)[tid]       = rb;
        __syncthreads();

        if (k0 + 32 < K) gload(k0 + 32);           // overlap next-tile loads with compute
        if (k0 + 64 < K) {                          // prefetch tile after next into L2/WGP$
            __builtin_prefetch(A + (size_t)(m0 + rowA) * K + k0 + 64, 0, 0);
            __builtin_prefetch(W + (size_t)(k0 + 64 + rowB) * N + n0, 0, 0);
        }

        FragAB a, bf[4];
#pragma unroll
        for (int j = 0; j < 8; ++j) {
            int idx8 = (j & 3) + ((j >> 2) << 3) + (half << 2);   // packed-pair index per ISA A layout
            a.u[j] = ldsA[(w * 16 + l16) * 16 + idx8];
        }
#pragma unroll
        for (int nt = 0; nt < 4; ++nt)
#pragma unroll
            for (int j = 0; j < 8; ++j) {  // lane carries N column; VGPR j carries K pair
                int kk = ((j & 3) << 1) + ((j >> 2) << 4) + (half << 3);
                const u16* lb = (const u16*)ldsB;
                bf[nt].u[j] = (u32)lb[kk * 64 + nt * 16 + l16] | ((u32)lb[(kk + 1) * 64 + nt * 16 + l16] << 16);
            }
#pragma unroll
        for (int nt = 0; nt < 4; ++nt)
            acc[nt] = __builtin_amdgcn_wmma_f32_16x16x32_bf16(false, a.v, false, bf[nt].v,
                                                              (short)0, acc[nt], false, false);
        __syncthreads();
    }

    int rowoff = w * 16 + (half << 3);
#pragma unroll
    for (int nt = 0; nt < 4; ++nt) {
        int col = n0 + nt * 16 + l16;
        float bi = bias[col];
#pragma unroll
        for (int v = 0; v < 8; ++v) {
            int row = m0 + rowoff + v;
            float val = acc[nt][v] + bi;
            if (RELU) val = fmaxf(val, 0.f);
            size_t off = (size_t)row * N + col;
            if (HAS_RESID) val += resid[off];
            if (OUTF) outF[off] = val;
            if (OUTB) outB[off] = f2bf(val);
        }
    }
}

// ---------------------------------------------------------------- flash attention (WMMA)
// Block: 128 thr (4 waves), 64 queries per block, one (b,h) per blockIdx.y.
__global__ __launch_bounds__(128) void attn_kernel(const u16* __restrict__ Qb, const u16* __restrict__ Kb,
                                                   const u16* __restrict__ Vb,
                                                   const float* __restrict__ tf, const float* __restrict__ pm,
                                                   const float* __restrict__ alpha, int layer,
                                                   u16* __restrict__ outB) {
    __shared__ __align__(16) u32 ldsK[64 * 16];     // 64 keys x 32 dh bf16
    __shared__ __align__(16) u32 ldsV[64 * 16];
    __shared__ __align__(16) u16 ldsP[4 * 16 * 64]; // per-wave P tiles
    int tid = threadIdx.x, w = tid >> 5, lane = tid & 31;
    int half = lane >> 4, l16 = lane & 15;
    int bh = blockIdx.y; int b = bh >> 3; int h = bh & 7;
    int q0 = blockIdx.x * 64;
    float al = alpha[layer];
    const float scale = 0.17677669529663687f;       // 1/sqrt(32)

    // Q fragment (A layout), held in registers for all key chunks
    FragAB qf;
    {
        int q = q0 + w * 16 + l16;
        const u32* qp = (const u32*)Qb + ((((size_t)(b * S_ + q)) * D_ + h * DH_) >> 1);
#pragma unroll
        for (int j = 0; j < 8; ++j) {
            int idx8 = (j & 3) + ((j >> 2) << 3) + (half << 2);
            qf.u[j] = qp[idx8];
        }
    }

    float m8[8], s8[8]; v8f o0, o1;
#pragma unroll
    for (int v = 0; v < 8; ++v) { m8[v] = -3e38f; s8[v] = 0.f; o0[v] = 0.f; o1[v] = 0.f; }

    for (int kc = 0; kc < S_ / 64; ++kc) {
        int kb0 = kc * 64;
#pragma unroll
        for (int it = 0; it < 2; ++it) {
            int idx = tid + it * 128;
            int row = idx >> 2, q4 = idx & 3;
            const u16* kp = Kb + (size_t)(b * S_ + kb0 + row) * D_ + h * DH_;
            const u16* vp = Vb + (size_t)(b * S_ + kb0 + row) * D_ + h * DH_;
            ((uint4*)ldsK)[idx] = *((const uint4*)kp + q4);
            ((uint4*)ldsV)[idx] = *((const uint4*)vp + q4);
        }
        __syncthreads();

        if (kc + 1 < S_ / 64) {                    // prefetch next chunk (global_prefetch_b8)
            int row = tid >> 1;
            __builtin_prefetch(Kb + (size_t)(b * S_ + kb0 + 64 + row) * D_ + h * DH_, 0, 0);
            __builtin_prefetch(Vb + (size_t)(b * S_ + kb0 + 64 + row) * D_ + h * DH_, 0, 0);
        }

        // scores: 4 tiles of 16 keys, K-dim = DH = 32; load all fragments, then WMMA back-to-back
        const u16* ldsK16 = (const u16*)ldsK;
        FragAB kf[4];
#pragma unroll
        for (int nt = 0; nt < 4; ++nt)
#pragma unroll
            for (int j = 0; j < 8; ++j) {          // lane carries key; VGPR j carries dh pair
                int idx8 = (j & 3) + ((j >> 2) << 3) + (half << 2);
                kf[nt].u[j] = ldsK[(nt * 16 + l16) * 16 + idx8];
            }
        v8f sc[4];
#pragma unroll
        for (int nt = 0; nt < 4; ++nt) {
            v8f z;
#pragma unroll
            for (int v = 0; v < 8; ++v) z[v] = 0.f;
            sc[nt] = __builtin_amdgcn_wmma_f32_16x16x32_bf16(false, qf.v, false, kf[nt].v,
                                                             (short)0, z, false, false);
        }
#pragma unroll
        for (int nt = 0; nt < 4; ++nt) {
            int key = kb0 + nt * 16 + l16;
            float addv = (pm[b * S_ + key] != 0.f) ? -1e30f : tf[b * S_ + key] * al;
#pragma unroll
            for (int v = 0; v < 8; ++v) sc[nt][v] = sc[nt][v] * scale + addv;
        }

        // online softmax (rows live across the 16-lane half; reduce with xor 1,2,4,8)
        float cm[8];
#pragma unroll
        for (int v = 0; v < 8; ++v)
            cm[v] = fmaxf(fmaxf(sc[0][v], sc[1][v]), fmaxf(sc[2][v], sc[3][v]));
        for (int o = 1; o < 16; o <<= 1)
#pragma unroll
            for (int v = 0; v < 8; ++v) cm[v] = fmaxf(cm[v], __shfl_xor(cm[v], o, 32));
        float corr[8];
#pragma unroll
        for (int v = 0; v < 8; ++v) {
            float nm = fmaxf(m8[v], cm[v]);
            corr[v] = __expf(m8[v] - nm);
            m8[v] = nm;
        }
        float rs[8];
#pragma unroll
        for (int v = 0; v < 8; ++v) rs[v] = 0.f;
#pragma unroll
        for (int nt = 0; nt < 4; ++nt)
#pragma unroll
            for (int v = 0; v < 8; ++v) { float p = __expf(sc[nt][v] - m8[v]); sc[nt][v] = p; rs[v] += p; }
        for (int o = 1; o < 16; o <<= 1)
#pragma unroll
            for (int v = 0; v < 8; ++v) rs[v] += __shfl_xor(rs[v], o, 32);
#pragma unroll
        for (int v = 0; v < 8; ++v) { s8[v] = s8[v] * corr[v] + rs[v]; o0[v] *= corr[v]; o1[v] *= corr[v]; }

        // P (C layout) -> per-wave LDS tile -> A-layout fragments
        u16* pw = ldsP + w * 1024;
#pragma unroll
        for (int nt = 0; nt < 4; ++nt) {
            int col = nt * 16 + l16;
#pragma unroll
            for (int v = 0; v < 8; ++v) {
                int r = v + (half << 3);
                pw[r * 64 + col] = f2bf(sc[nt][v]);
            }
        }
        const u32* pu = (const u32*)ldsP + w * 512;
#pragma unroll
        for (int kc2 = 0; kc2 < 2; ++kc2) {
            FragAB pf;
#pragma unroll
            for (int j = 0; j < 8; ++j) {
                int idx8 = (j & 3) + ((j >> 2) << 3) + (half << 2);
                pf.u[j] = pu[l16 * 32 + kc2 * 16 + idx8];
            }
            FragAB vf0, vf1;
            const u16* ldsV16 = (const u16*)ldsV;  // lane carries dh column
#pragma unroll
            for (int j = 0; j < 8; ++j) {
                int kk = kc2 * 32 + ((j & 3) << 1) + ((j >> 2) << 4) + (half << 3);
                vf0.u[j] = (u32)ldsV16[kk * 32 + l16] | ((u32)ldsV16[(kk + 1) * 32 + l16] << 16);        // dh 0..15
                vf1.u[j] = (u32)ldsV16[kk * 32 + 16 + l16] | ((u32)ldsV16[(kk + 1) * 32 + 16 + l16] << 16);    // dh 16..31
            }
            o0 = __builtin_amdgcn_wmma_f32_16x16x32_bf16(false, pf.v, false, vf0.v, (short)0, o0, false, false);
            o1 = __builtin_amdgcn_wmma_f32_16x16x32_bf16(false, pf.v, false, vf1.v, (short)0, o1, false, false);
        }
        __syncthreads();
    }

    int qrow = q0 + w * 16 + (half << 3);
#pragma unroll
    for (int v = 0; v < 8; ++v) {
        float inv = 1.f / s8[v];
        int q = qrow + v;
        u16* orow = outB + (size_t)(b * S_ + q) * D_ + h * DH_;
        orow[l16]      = f2bf(o0[v] * inv);
        orow[16 + l16] = f2bf(o1[v] * inv);
    }
}

// ---------------------------------------------------------------- final LN (CLS rows only) + classifier
__global__ __launch_bounds__(256) void cls_kernel(const float* __restrict__ x, const float* __restrict__ gf,
                                                  const float* __restrict__ bfp, const float* __restrict__ Wc,
                                                  const float* __restrict__ bc, float* __restrict__ out) {
    int w = threadIdx.x >> 5, lane = threadIdx.x & 31;  // w = batch index (8 waves)
    const float* xr = x + (size_t)w * S_ * D_;          // row (b, s=0)
    float r[8], s = 0.f;
#pragma unroll
    for (int i = 0; i < 8; ++i) { r[i] = xr[lane + 32 * i]; s += r[i]; }
    for (int o = 16; o > 0; o >>= 1) s += __shfl_xor(s, o, 32);
    float m = s * (1.f / D_);
    float s2 = 0.f;
#pragma unroll
    for (int i = 0; i < 8; ++i) { float d = r[i] - m; s2 += d * d; }
    for (int o = 16; o > 0; o >>= 1) s2 += __shfl_xor(s2, o, 32);
    float inv = rsqrtf(s2 * (1.f / D_) + 1e-5f);
    float acc[4] = {0.f, 0.f, 0.f, 0.f};
#pragma unroll
    for (int i = 0; i < 8; ++i) {
        int d = lane + 32 * i;
        float n = (r[i] - m) * inv * gf[d] + bfp[d];
#pragma unroll
        for (int c = 0; c < 4; ++c) acc[c] += n * Wc[d * C_ + c];
    }
    for (int o = 16; o > 0; o >>= 1)
#pragma unroll
        for (int c = 0; c < 4; ++c) acc[c] += __shfl_xor(acc[c], o, 32);
    if (lane == 0)
#pragma unroll
        for (int c = 0; c < 4; ++c) out[w * C_ + c] = acc[c] + bc[c];
}

// ----------------------------------------------------------------
extern "C" void kernel_launch(void* const* d_in, const int* in_sizes, int n_in,
                              void* d_out, int out_size, void* d_ws, size_t ws_size,
                              hipStream_t stream) {
    const int*   input_ids = (const int*)d_in[0];
    const int*   ngram_ids = (const int*)d_in[1];
    const float* tfidf     = (const float*)d_in[2];
    const float* emb       = (const float*)d_in[3];
    const float* pos       = (const float*)d_in[4];
    const float* bucket    = (const float*)d_in[5];
    const float* Wq = (const float*)d_in[6],  *bq = (const float*)d_in[7];
    const float* Wk = (const float*)d_in[8],  *bk = (const float*)d_in[9];
    const float* Wv = (const float*)d_in[10], *bv = (const float*)d_in[11];
    const float* Wo = (const float*)d_in[12], *bo = (const float*)d_in[13];
    const float* g1 = (const float*)d_in[14], *be1 = (const float*)d_in[15];
    const float* g2 = (const float*)d_in[16], *be2 = (const float*)d_in[17];
    const float* W1 = (const float*)d_in[18], *b1 = (const float*)d_in[19];
    const float* W2 = (const float*)d_in[20], *b2 = (const float*)d_in[21];
    const float* alpha = (const float*)d_in[22];
    const float* gf = (const float*)d_in[23], *bff = (const float*)d_in[24];
    const float* Wc = (const float*)d_in[25], *bc = (const float*)d_in[26];
    float* out = (float*)d_out;

    // workspace carve-out
    char* wsb = (char*)d_ws;
    size_t off = 0;
    auto alloc = [&](size_t bytes) -> void* {
        void* p = wsb + off;
        off += (bytes + 255) & ~(size_t)255;
        return p;
    };
    float* x   = (float*)alloc((size_t)BS_ * D_ * 4);
    u16* x2b   = (u16*)alloc((size_t)BS_ * D_ * 2);
    u16* qb    = (u16*)alloc((size_t)BS_ * D_ * 2);
    u16* kb    = (u16*)alloc((size_t)BS_ * D_ * 2);
    u16* vb    = (u16*)alloc((size_t)BS_ * D_ * 2);
    u16* ab    = (u16*)alloc((size_t)BS_ * D_ * 2);
    u16* hb    = (u16*)alloc((size_t)BS_ * FF_ * 2);
    u16* WqB   = (u16*)alloc((size_t)L_ * D_ * D_ * 2);
    u16* WkB   = (u16*)alloc((size_t)L_ * D_ * D_ * 2);
    u16* WvB   = (u16*)alloc((size_t)L_ * D_ * D_ * 2);
    u16* WoB   = (u16*)alloc((size_t)L_ * D_ * D_ * 2);
    u16* W1B   = (u16*)alloc((size_t)L_ * D_ * FF_ * 2);
    u16* W2B   = (u16*)alloc((size_t)L_ * FF_ * D_ * 2);
    float* tf  = (float*)alloc((size_t)BS_ * 4);
    float* pm  = (float*)alloc((size_t)BS_ * 4);

    // weight conversion to bf16 (deterministic, every call)
    const int nDD = L_ * D_ * D_, nDF = L_ * D_ * FF_;
    cvt_f32_bf16<<<(nDD + 255) / 256, 256, 0, stream>>>(Wq, WqB, nDD);
    cvt_f32_bf16<<<(nDD + 255) / 256, 256, 0, stream>>>(Wk, WkB, nDD);
    cvt_f32_bf16<<<(nDD + 255) / 256, 256, 0, stream>>>(Wv, WvB, nDD);
    cvt_f32_bf16<<<(nDD + 255) / 256, 256, 0, stream>>>(Wo, WoB, nDD);
    cvt_f32_bf16<<<(nDF + 255) / 256, 256, 0, stream>>>(W1, W1B, nDF);
    cvt_f32_bf16<<<(nDF + 255) / 256, 256, 0, stream>>>(W2, W2B, nDF);

    // embeddings + key bias/mask
    embed_kernel<<<BS_, D_, 0, stream>>>(input_ids, ngram_ids, tfidf, emb, pos, bucket, x, tf, pm);

    dim3 blk256(256), blk128(128);
    dim3 gridN256(D_ / 64, BS_ / 128);   // N=256 GEMMs
    dim3 gridN1024(FF_ / 64, BS_ / 128); // N=1024 GEMM
    dim3 gridAttn(S_ / 64, B_ * H_);

    for (int l = 0; l < L_; ++l) {
        // LN1 -> x2b (bf16)
        ln_bf16_kernel<<<BS_ / 8, blk256, 0, stream>>>(x, g1 + l * D_, be1 + l * D_, x2b);
        // Q, K, V projections (bf16 out):       RELU=0, RESID=0, OUTF=0, OUTB=1
        gemm_wmma<0, 0, 0, 1><<<gridN256, blk256, 0, stream>>>(x2b, WqB + (size_t)l * D_ * D_, bq + l * D_,
                                                               nullptr, nullptr, qb, BS_, D_, D_);
        gemm_wmma<0, 0, 0, 1><<<gridN256, blk256, 0, stream>>>(x2b, WkB + (size_t)l * D_ * D_, bk + l * D_,
                                                               nullptr, nullptr, kb, BS_, D_, D_);
        gemm_wmma<0, 0, 0, 1><<<gridN256, blk256, 0, stream>>>(x2b, WvB + (size_t)l * D_ * D_, bv + l * D_,
                                                               nullptr, nullptr, vb, BS_, D_, D_);
        // fused attention -> ab (bf16)
        attn_kernel<<<gridAttn, blk128, 0, stream>>>(qb, kb, vb, tf, pm, alpha, l, ab);
        // output projection + residual -> x:    RELU=0, RESID=1, OUTF=1, OUTB=0
        gemm_wmma<0, 1, 1, 0><<<gridN256, blk256, 0, stream>>>(ab, WoB + (size_t)l * D_ * D_, bo + l * D_,
                                                               x, x, nullptr, BS_, D_, D_);
        // LN2 -> x2b
        ln_bf16_kernel<<<BS_ / 8, blk256, 0, stream>>>(x, g2 + l * D_, be2 + l * D_, x2b);
        // FFN up + ReLU -> hb (bf16):           RELU=1, RESID=0, OUTF=0, OUTB=1
        gemm_wmma<1, 0, 0, 1><<<gridN1024, blk256, 0, stream>>>(x2b, W1B + (size_t)l * D_ * FF_, b1 + l * FF_,
                                                                nullptr, nullptr, hb, BS_, FF_, D_);
        // FFN down + residual -> x
        gemm_wmma<0, 1, 1, 0><<<gridN256, blk256, 0, stream>>>(hb, W2B + (size_t)l * FF_ * D_, b2 + l * D_,
                                                               x, x, nullptr, BS_, D_, FF_);
    }

    // final LN (CLS rows) + classifier
    cls_kernel<<<1, blk256, 0, stream>>>(x, gf, bff, Wc, bc, out);
}
